// Net_30992484008405
// MI455X (gfx1250) — matrix operand order repaired and checked by reading, not compiled, not run on hardware
//
#include <hip/hip_runtime.h>
#include <hip/hip_bf16.h>
#include <stdint.h>

#define NN 8192     // nodes
#define NE 32768    // edges
#define NB 256      // graphs
#define EPS 1e-5f

typedef __attribute__((ext_vector_type(16))) _Float16 v16h;
typedef __attribute__((ext_vector_type(8)))  float    v8f;

union Frag16 {
  v16h h;
  unsigned int u[8];
  uint4 u4[2];
};

__device__ __forceinline__ float lrelu(float x) { return x > 0.0f ? x : 0.01f * x; }
__device__ __forceinline__ float sigm(float x)  { return 1.0f / (1.0f + expf(-x)); }

// ---------------------------------------------------------------- utilities
__global__ void k_zero(float* p, int n) {
  int t = blockIdx.x * blockDim.x + threadIdx.x;
  if (t < n) p[t] = 0.0f;
}

// out0 = lrelu(bn0(x @ lin0_w^T))  -> cur [NN,64]
__global__ void k_init_nodes(const float* __restrict__ x, const float* __restrict__ w,
                             const float* __restrict__ g, const float* __restrict__ b,
                             const float* __restrict__ m, const float* __restrict__ v,
                             float* __restrict__ cur) {
  int t = blockIdx.x * blockDim.x + threadIdx.x;
  if (t >= NN * 64) return;
  int n = t >> 6, o = t & 63;
  float s = 0.0f;
  #pragma unroll
  for (int k = 0; k < 16; ++k) s += x[n * 16 + k] * w[o * 16 + k];
  float rs = rsqrtf(v[o] + EPS);
  cur[t] = lrelu(g[o] * (s - m[o]) * rs + b[o]);
}

// eh16 = fp16(lrelu(bn1(edge_attr @ enn_w1^T)))  [NE,128]
__global__ void k_eh(const float* __restrict__ ea, const float* __restrict__ w1,
                     const float* __restrict__ g, const float* __restrict__ b,
                     const float* __restrict__ m, const float* __restrict__ v,
                     _Float16* __restrict__ eh16) {
  int t = blockIdx.x * blockDim.x + threadIdx.x;
  if (t >= NE * 128) return;
  int e = t >> 7, j = t & 127;
  float s = 0.0f;
  #pragma unroll
  for (int k = 0; k < 8; ++k) s += ea[e * 8 + k] * w1[j * 8 + k];
  float rs = rsqrtf(v[j] + EPS);
  eh16[t] = (_Float16)lrelu(g[j] * (s - m[j]) * rs + b[j]);
}

// Cmat[io] = bn2_b - bn2_g*bn2_m*rsqrt(v+eps)   (additive BN term of Wmat)
__global__ void k_cmat(const float* g, const float* b, const float* m, const float* v,
                       float* C) {
  int t = blockIdx.x * blockDim.x + threadIdx.x;
  if (t < 4096) { float rs = rsqrtf(v[t] + EPS); C[t] = b[t] - g[t] * m[t] * rs; }
}

// Pack (bn2_scale * enn_w2) into WMMA-B fragment order (fp16).
// Fragment id f = i*16 + osub*4 + ks  (i: Wmat input row, osub: 16-wide output
// sub-tile, ks: 32-wide K step). B element (Klocal, Nlocal=lane&15):
// Klocal = 2v + h + 16*(lane>>4). Stored so lane reads 2 coalesced uint4's.
__global__ void k_repack(const float* __restrict__ w2, const float* __restrict__ g,
                         const float* __restrict__ v, unsigned int* __restrict__ Bp) {
  int f = blockIdx.x;          // 0..1023
  int lane = threadIdx.x;      // 0..31
  int i = f >> 4;
  int osub = (f >> 2) & 3;
  int ks = f & 3;
  int lo = lane & 15, hi = lane >> 4;
  int row = i * 64 + osub * 16 + lo;              // row of enn_w2 (i*64+o)
  float a = g[row] * rsqrtf(v[row] + EPS);        // fold BN scale into B
  #pragma unroll
  for (int vv = 0; vv < 8; ++vv) {
    int k = ks * 32 + 2 * vv + 16 * hi;
    union { _Float16 h[2]; unsigned int u; } p;
    p.h[0] = (_Float16)(a * w2[row * 128 + k]);
    p.h[1] = (_Float16)(a * w2[row * 128 + k + 1]);
    Bp[((f * 2 + (vv >> 2)) * 32 + lane) * 4 + (vv & 3)] = p.u;
  }
}

// deg[n] = number of incoming edges (exact: float adds of 1.0)
__global__ void k_deg(const int* __restrict__ ei, float* __restrict__ deg) {
  int t = blockIdx.x * blockDim.x + threadIdx.x;
  if (t < NE) atomicAdd(&deg[ei[NE + t]], 1.0f);
}

// Fused per-edge:  msg_e = out[src_e]^T * bn2(eh_e @ w2^T)  -> atomic into agg[dst_e]
// One wave handles 32 edges (two 16-edge WMMA tiles) so each B-fragment load
// (edge-invariant weights) feeds TWO v_wmma_f32_16x16x32_f16 ops: 32 FLOP/byte
// from L2 instead of 16, and 1 b128 load per WMMA instead of 2.
__global__ void __launch_bounds__(32)
k_msg(const int* __restrict__ ei, const float* __restrict__ cur,
      const _Float16* __restrict__ eh16, const uint4* __restrict__ Bp4,
      const float* __restrict__ Cmat, float* __restrict__ agg) {
  __shared__ int s_src[32];
  __shared__ int s_dst[32];
  __shared__ float s_out[32 * 64];
  const int lane = threadIdx.x;
  const int e0 = blockIdx.x * 32;
  const int lo = lane & 15, hi = lane >> 4;

  s_src[lane] = ei[e0 + lane];
  s_dst[lane] = ei[NE + e0 + lane];
  __syncthreads();
  for (int idx = lane; idx < 32 * 64; idx += 32)
    s_out[idx] = cur[s_src[idx >> 6] * 64 + (idx & 63)];
  __syncthreads();

  // A fragments: eh tiles [16 edges x 32 K] x2, fp16, contiguous-K layout (same as B pack):
  // row M = lane&15; K = 2v + 16*(lane>>4) (+32 per k-step)
  Frag16 afrA[4], afrB[4];
  const _Float16* ehrowA = eh16 + (e0 + lo) * 128;
  const _Float16* ehrowB = eh16 + (e0 + 16 + lo) * 128;
  #pragma unroll
  for (int ks = 0; ks < 4; ++ks) {
    #pragma unroll
    for (int vv = 0; vv < 8; ++vv) {
      int k = ks * 32 + 2 * vv + 16 * hi;
      afrA[ks].u[vv] = *(const unsigned int*)(ehrowA + k);
      afrB[ks].u[vv] = *(const unsigned int*)(ehrowB + k);
    }
  }

  // message accumulators in WMMA D layout: [osub][reg r] -> edge r+8*hi, o=osub*16+lo
  float maccA[4][8], maccB[4][8];
  #pragma unroll
  for (int a = 0; a < 4; ++a)
    #pragma unroll
    for (int r = 0; r < 8; ++r) { maccA[a][r] = 0.0f; maccB[a][r] = 0.0f; }

  for (int i = 0; i < 64; ++i) {
    float svA[8], svB[8];
    #pragma unroll
    for (int r = 0; r < 8; ++r) {
      svA[r] = s_out[(r + 8 * hi) * 64 + i];        // out[src][i], edges 0..15
      svB[r] = s_out[(16 + r + 8 * hi) * 64 + i];   // edges 16..31
    }
    #pragma unroll
    for (int osub = 0; osub < 4; ++osub) {
      v8f accA = {0.f, 0.f, 0.f, 0.f, 0.f, 0.f, 0.f, 0.f};
      v8f accB = {0.f, 0.f, 0.f, 0.f, 0.f, 0.f, 0.f, 0.f};
      #pragma unroll
      for (int ks = 0; ks < 4; ++ks) {
        const int f = i * 16 + osub * 4 + ks;
        Frag16 bf;
        bf.u4[0] = Bp4[f * 64 + lane];
        bf.u4[1] = Bp4[f * 64 + 32 + lane];
        accA = __builtin_amdgcn_wmma_f32_16x16x32_f16(false, afrA[ks].h, false, bf.h,
                                                      (short)0, accA, false, false);
        accB = __builtin_amdgcn_wmma_f32_16x16x32_f16(false, afrB[ks].h, false, bf.h,
                                                      (short)0, accB, false, false);
      }
      const float c = Cmat[i * 64 + osub * 16 + lo];   // BN additive term
      #pragma unroll
      for (int r = 0; r < 8; ++r) {
        maccA[osub][r] += svA[r] * (accA[r] + c);
        maccB[osub][r] += svB[r] * (accB[r] + c);
      }
    }
  }

  #pragma unroll
  for (int osub = 0; osub < 4; ++osub)
    #pragma unroll
    for (int r = 0; r < 8; ++r) {
      const int elA = r + 8 * hi;
      const int elB = 16 + r + 8 * hi;
      atomicAdd(&agg[s_dst[elA] * 64 + osub * 16 + lo], maccA[osub][r]);
      atomicAdd(&agg[s_dst[elB] * 64 + osub * 16 + lo], maccB[osub][r]);
    }
}

// m = lrelu(agg/deg + cur@root_w^T + root_b);  GRU update of cur (in place per node)
__global__ void __launch_bounds__(64)
k_node(float* __restrict__ cur, const float* __restrict__ agg,
       const float* __restrict__ deg,
       const float* __restrict__ root_w, const float* __restrict__ root_b,
       const float* __restrict__ wih, const float* __restrict__ whh,
       const float* __restrict__ bih, const float* __restrict__ bhh) {
  __shared__ float s_h[64];
  __shared__ float s_m[64];
  const int n = blockIdx.x;
  const int t = threadIdx.x;
  s_h[t] = cur[n * 64 + t];
  __syncthreads();
  float d = fmaxf(deg[n], 1.0f);
  float acc = root_b[t];
  for (int k = 0; k < 64; ++k) acc += s_h[k] * root_w[t * 64 + k];
  s_m[t] = lrelu(agg[n * 64 + t] / d + acc);
  __syncthreads();
  float gi0 = bih[t], gi1 = bih[t + 64], gi2 = bih[t + 128];
  float gh0 = bhh[t], gh1 = bhh[t + 64], gh2 = bhh[t + 128];
  for (int k = 0; k < 64; ++k) {
    float mk = s_m[k], hk = s_h[k];
    gi0 += mk * wih[t * 64 + k];
    gi1 += mk * wih[(t + 64) * 64 + k];
    gi2 += mk * wih[(t + 128) * 64 + k];
    gh0 += hk * whh[t * 64 + k];
    gh1 += hk * whh[(t + 64) * 64 + k];
    gh2 += hk * whh[(t + 128) * 64 + k];
  }
  float r  = sigm(gi0 + gh0);
  float z  = sigm(gi1 + gh1);
  float nn = tanhf(gi2 + r * gh2);
  cur[n * 64 + t] = (1.0f - z) * nn + z * s_h[t];
}

// segment boundaries of sorted `batch` via binary search; seg[NB] = NN
__global__ void k_seg(const int* __restrict__ batch, int* __restrict__ seg) {
  int b = threadIdx.x;
  int lo = 0, hi = NN;
  while (lo < hi) { int mid = (lo + hi) >> 1; if (batch[mid] < b) lo = mid + 1; else hi = mid; }
  seg[b] = lo;
  if (b == 0) seg[NB] = NN;
}

// Set2Set LSTM cell: gates = q_star@wih^T + hx@whh^T + biases; update hx,cx; q part of q_star
__global__ void __launch_bounds__(64)
k_lstm(float* __restrict__ qstar, float* __restrict__ hx, float* __restrict__ cx,
       const float* __restrict__ wih, const float* __restrict__ whh,
       const float* __restrict__ bih, const float* __restrict__ bhh) {
  __shared__ float s_q[128];
  __shared__ float s_h[64];
  const int b = blockIdx.x;
  const int t = threadIdx.x;
  s_q[t] = qstar[b * 128 + t];
  s_q[t + 64] = qstar[b * 128 + 64 + t];
  s_h[t] = hx[b * 64 + t];
  __syncthreads();
  float gate[4];
  #pragma unroll
  for (int q = 0; q < 4; ++q) {
    int j = t + 64 * q;
    float a = bih[j] + bhh[j];
    for (int k = 0; k < 128; ++k) a += s_q[k] * wih[j * 128 + k];
    for (int k = 0; k < 64; ++k)  a += s_h[k] * whh[j * 64 + k];
    gate[q] = a;
  }
  float c = sigm(gate[1]) * cx[b * 64 + t] + sigm(gate[0]) * tanhf(gate[2]);
  float h = sigm(gate[3]) * tanhf(c);
  cx[b * 64 + t] = c;
  hx[b * 64 + t] = h;
  qstar[b * 128 + t] = h;       // q half of q_star
}

// segment softmax attention + readout: q_star[b][64:128] = sum_n a_n * cur[n]
__global__ void __launch_bounds__(64)
k_attn(const float* __restrict__ cur, const float* __restrict__ hx,
       float* __restrict__ qstar, const int* __restrict__ seg,
       float* __restrict__ eN) {
  __shared__ float s_q[64];
  __shared__ float red[64];
  const int b = blockIdx.x;
  const int t = threadIdx.x;
  s_q[t] = hx[b * 64 + t];
  __syncthreads();
  const int st = seg[b], en = seg[b + 1];
  float lm = -INFINITY;
  for (int n = st + t; n < en; n += 64) {
    float e = 0.0f;
    for (int o = 0; o < 64; ++o) e += cur[n * 64 + o] * s_q[o];
    eN[n] = e;
    lm = fmaxf(lm, e);
  }
  red[t] = lm; __syncthreads();
  for (int s = 32; s > 0; s >>= 1) { if (t < s) red[t] = fmaxf(red[t], red[t + s]); __syncthreads(); }
  float emax = red[0];
  if (!isfinite(emax)) emax = 0.0f;
  __syncthreads();
  float ls = 0.0f;
  for (int n = st + t; n < en; n += 64) ls += expf(eN[n] - emax);
  red[t] = ls; __syncthreads();
  for (int s = 32; s > 0; s >>= 1) { if (t < s) red[t] += red[t + s]; __syncthreads(); }
  float denom = fmaxf(red[0], 1e-16f);
  float racc = 0.0f;
  for (int n = st; n < en; ++n) racc += expf(eN[n] - emax) * cur[n * 64 + t];
  qstar[b * 128 + 64 + t] = racc / denom;
}

// MLP head: lrelu(q_star@lin1^T+b) -> lrelu(@lin2^T+b) -> @linf^T+b -> out[b]
__global__ void __launch_bounds__(64)
k_final(const float* __restrict__ qstar,
        const float* __restrict__ w1, const float* __restrict__ b1,
        const float* __restrict__ w2, const float* __restrict__ b2,
        const float* __restrict__ wf, const float* __restrict__ bf,
        float* __restrict__ out) {
  __shared__ float s_qs[128];
  __shared__ float s_y1[64];
  __shared__ float s_y2[32];
  const int b = blockIdx.x;
  const int t = threadIdx.x;
  s_qs[t] = qstar[b * 128 + t];
  s_qs[t + 64] = qstar[b * 128 + 64 + t];
  __syncthreads();
  float a = b1[t];
  for (int k = 0; k < 128; ++k) a += s_qs[k] * w1[t * 128 + k];
  s_y1[t] = lrelu(a);
  __syncthreads();
  if (t < 32) {
    float a2 = b2[t];
    for (int k = 0; k < 64; ++k) a2 += s_y1[k] * w2[t * 64 + k];
    s_y2[t] = lrelu(a2);
  }
  __syncthreads();
  if (t == 0) {
    float a3 = bf[0];
    for (int k = 0; k < 32; ++k) a3 += s_y2[k] * wf[k];
    out[b] = a3;
  }
}

// ---------------------------------------------------------------- launch
extern "C" void kernel_launch(void* const* d_in, const int* in_sizes, int n_in,
                              void* d_out, int out_size, void* d_ws, size_t ws_size,
                              hipStream_t stream) {
  (void)in_sizes; (void)n_in; (void)out_size; (void)ws_size;
  const float* x      = (const float*)d_in[0];
  const int*   ei     = (const int*)d_in[1];
  const float* ea     = (const float*)d_in[2];
  const int*   batch  = (const int*)d_in[3];
  const float* lin0_w = (const float*)d_in[4];
  const float* bn0_g  = (const float*)d_in[5];
  const float* bn0_b  = (const float*)d_in[6];
  const float* bn0_m  = (const float*)d_in[7];
  const float* bn0_v  = (const float*)d_in[8];
  const float* w1     = (const float*)d_in[9];
  const float* bn1_g  = (const float*)d_in[10];
  const float* bn1_b  = (const float*)d_in[11];
  const float* bn1_m  = (const float*)d_in[12];
  const float* bn1_v  = (const float*)d_in[13];
  const float* w2     = (const float*)d_in[14];
  const float* bn2_g  = (const float*)d_in[15];
  const float* bn2_b  = (const float*)d_in[16];
  const float* bn2_m  = (const float*)d_in[17];
  const float* bn2_v  = (const float*)d_in[18];
  const float* root_w = (const float*)d_in[19];
  const float* root_b = (const float*)d_in[20];
  const float* gwih   = (const float*)d_in[21];
  const float* gwhh   = (const float*)d_in[22];
  const float* gbih   = (const float*)d_in[23];
  const float* gbhh   = (const float*)d_in[24];
  const float* lwih   = (const float*)d_in[25];
  const float* lwhh   = (const float*)d_in[26];
  const float* lbih   = (const float*)d_in[27];
  const float* lbhh   = (const float*)d_in[28];
  const float* lin1_w = (const float*)d_in[29];
  const float* lin1_b = (const float*)d_in[30];
  const float* lin2_w = (const float*)d_in[31];
  const float* lin2_b = (const float*)d_in[32];
  const float* linf_w = (const float*)d_in[33];
  const float* linf_b = (const float*)d_in[34];
  float* outp = (float*)d_out;

  char* ws = (char*)d_ws;
  size_t off = 0;
  auto alloc = [&](size_t bytes) -> void* {
    off = (off + 255) & ~(size_t)255;
    void* p = ws + off;
    off += bytes;
    return p;
  };
  float*        cur   = (float*)alloc((size_t)NN * 64 * 4);
  float*        agg   = (float*)alloc((size_t)NN * 64 * 4);
  float*        deg   = (float*)alloc((size_t)NN * 4);
  float*        Cmat  = (float*)alloc(4096 * 4);
  float*        eN    = (float*)alloc((size_t)NN * 4);
  int*          seg   = (int*)alloc((NB + 1) * 4);
  float*        qstar = (float*)alloc((size_t)NB * 128 * 4);   // qstar,hx,cx contiguous
  float*        hx    = (float*)alloc((size_t)NB * 64 * 4);
  float*        cx    = (float*)alloc((size_t)NB * 64 * 4);
  _Float16*     eh16  = (_Float16*)alloc((size_t)NE * 128 * 2);
  unsigned int* Bp    = (unsigned int*)alloc((size_t)1024 * 64 * 16);  // 1024 frags * 64 uint4

  // one-time precompute (weights + edge network, step-invariant)
  k_init_nodes<<<(NN * 64 + 255) / 256, 256, 0, stream>>>(x, lin0_w, bn0_g, bn0_b, bn0_m, bn0_v, cur);
  k_eh<<<(NE * 128 + 255) / 256, 256, 0, stream>>>(ea, w1, bn1_g, bn1_b, bn1_m, bn1_v, eh16);
  k_cmat<<<16, 256, 0, stream>>>(bn2_g, bn2_b, bn2_m, bn2_v, Cmat);
  k_repack<<<1024, 32, 0, stream>>>(w2, bn2_g, bn2_v, Bp);
  k_zero<<<(NN + 255) / 256, 256, 0, stream>>>(deg, NN);
  k_deg<<<(NE + 255) / 256, 256, 0, stream>>>(ei, deg);

  // 3 GRU-coupled NNConv message-passing steps
  for (int step = 0; step < 3; ++step) {
    k_zero<<<(NN * 64 + 255) / 256, 256, 0, stream>>>(agg, NN * 64);
    k_msg<<<NE / 32, 32, 0, stream>>>(ei, cur, eh16, (const uint4*)Bp, Cmat, agg);
    k_node<<<NN, 64, 0, stream>>>(cur, agg, deg, root_w, root_b, gwih, gwhh, gbih, gbhh);
  }

  // Set2Set
  k_zero<<<(NB * 256 + 255) / 256, 256, 0, stream>>>(qstar, NB * 256);  // qstar+hx+cx
  k_seg<<<1, NB, 0, stream>>>(batch, seg);
  for (int it = 0; it < 3; ++it) {
    k_lstm<<<NB, 64, 0, stream>>>(qstar, hx, cx, lwih, lwhh, lbih, lbhh);
    k_attn<<<NB, 64, 0, stream>>>(cur, hx, qstar, seg, eN);
  }

  k_final<<<NB, 64, 0, stream>>>(qstar, lin1_w, lin1_b, lin2_w, lin2_b, linf_w, linf_b, outp);
}
